// GCNNEncoder_8031588844010
// MI455X (gfx1250) — matrix operand order repaired and checked by reading, not compiled, not run on hardware
//
#include <hip/hip_runtime.h>
#include <hip/hip_bf16.h>

typedef __attribute__((ext_vector_type(16))) _Float16 v16h;
typedef __attribute__((ext_vector_type(8)))  _Float16 v8h;
typedef __attribute__((ext_vector_type(8)))  float    v8f;
typedef __attribute__((ext_vector_type(4)))  int      v4i;

#define NB    1024
#define NN    11
#define NF    512
#define NP    121
#define NPPAD 128

// ---------------------------------------------------------------------------
// CDNA5 async global->LDS staging (ASYNCcnt) with portable fallback.
// Probe-discovered signature: (v4i addrspace(1)*, v4i addrspace(3)*, i32, i32)
// ---------------------------------------------------------------------------
#if __has_builtin(__builtin_amdgcn_global_load_async_to_lds_b128) && \
    __has_builtin(__builtin_amdgcn_s_wait_asynccnt)
#define USE_ASYNC_LDS 1
typedef __attribute__((address_space(1))) v4i* gv4i_p;   // global v4i*
typedef __attribute__((address_space(3))) v4i* lv4i_p;   // LDS v4i*
#else
#define USE_ASYNC_LDS 0
#endif

// Copy a 16x512 f16 panel (16 KB) global -> LDS, 256 threads cooperating.
// 1024 chunks of 16 B; each thread moves 4 chunks.
__device__ inline void stage_panel(_Float16* __restrict__ dst,
                                   const _Float16* __restrict__ src, int tid) {
#if USE_ASYNC_LDS
#pragma unroll
  for (int c = 0; c < 4; ++c) {
    const int ci = tid + c * 256;
    __builtin_amdgcn_global_load_async_to_lds_b128(
        (gv4i_p)(src + ci * 8), (lv4i_p)(dst + ci * 8), 0, 0);
  }
  __builtin_amdgcn_s_wait_asynccnt(0);
#else
#pragma unroll
  for (int c = 0; c < 4; ++c) {
    const int ci = tid + c * 256;
    *(v8h*)(dst + ci * 8) = *(const v8h*)(src + ci * 8);
  }
#endif
}

// ---------------------------------------------------------------------------
// WMMA helpers (CDNA5 v_wmma_f32_16x16x32_f16, wave32)
// ---------------------------------------------------------------------------
__device__ inline v8f wmma_f16(v16h a, v16h b, v8f c) {
  // 8 args: (neg_a, A, neg_b, B, c_mod, C, reuse_a, reuse_b)
  return __builtin_amdgcn_wmma_f32_16x16x32_f16(false, a, false, b, (short)0, c,
                                                false, false);
}

// A fragment: 16x32 (MxK), src points at tile origin of row-major f16 matrix.
// Lane l<16: row=l, K = {0..15}; lane l>=16: row=l-16, K = {16..31}.
__device__ inline v16h load_a_frag(const _Float16* __restrict__ src, int ld, int lane) {
  const int row = lane & 15;
  const int kh  = lane >> 4;
  const _Float16* p = src + row * ld + kh * 16;
  v8h lo = *(const v8h*)(p);
  v8h hi = *(const v8h*)(p + 8);
  v16h a;
#pragma unroll
  for (int i = 0; i < 8; ++i) { a[i] = lo[i]; a[8 + i] = hi[i]; }
  return a;
}

// B fragment: 32x16 (KxN) where B[k][n] = srcT[n][k], srcT row-major with ld.
// Lane: col = l&15, K-range = (l>>4)*16 .. +15 -> one contiguous 32-byte load.
__device__ inline v16h load_bt_frag(const _Float16* __restrict__ srcT, int ld, int lane) {
  const int col  = lane & 15;
  const int koff = (lane >> 4) * 16;
  return *(const v16h*)(srcT + col * ld + koff);
}

// ---------------------------------------------------------------------------
// Small utility kernels
// ---------------------------------------------------------------------------
__global__ void cvt_f16_kernel(const float* __restrict__ src,
                               _Float16* __restrict__ dst, int n) {
  int i = blockIdx.x * blockDim.x + threadIdx.x;
  if (i < n) dst[i] = (_Float16)src[i];
}

__global__ void init_col_kernel(const float* __restrict__ nf,
                                float* __restrict__ hcol, int n) {
  int i = blockIdx.x * blockDim.x + threadIdx.x;
  if (i < n) {
    int b = i >> 9, f = i & (NF - 1);
    hcol[i] = nf[(size_t)b * NN * NF + 10 * NF + f];   // x0[:, :, zidx]
  }
}

__global__ void copy_col_kernel(const float* __restrict__ hcol,
                                float* __restrict__ out, int n) {
  int i = blockIdx.x * blockDim.x + threadIdx.x;
  if (i < n) out[i] = hcol[i];
}

// ---------------------------------------------------------------------------
// ADJ kernel: per batch b, adj = w_out . relu(W2 . relu(W1 . E + b1) + b2) + b_out
// E[f][p] = P[i(p)][f] * P[j(p)][f],  p = i*11 + j, padded to 128 columns.
// Weight panels (16x512 f16) are staged in LDS once per g-tile and shared by
// all 8 waves (8x cut in L2 weight traffic vs per-wave streaming).
// LDS: P 11x512 f16 (11.25K) + H1T 121x512 f16 (121K) + panel 16K + 0.6K
//    = 152 KB -> 2 workgroups per 320 KB WGP.
// Also emits a_z = softmax(adj row 10) and base_m = sum_{j<10} a_z[j]*nf[:,j,:].
// ---------------------------------------------------------------------------
__global__ __launch_bounds__(256) void adj_kernel(
    const float* __restrict__ nf,       // (B, 11, 512)
    const float* __restrict__ hcol,     // (B, 512)
    const _Float16* __restrict__ W1h,   // (512, 512) f16
    const _Float16* __restrict__ W2h,   // (512, 512) f16
    const float* __restrict__ b1,
    const float* __restrict__ b2,
    const float* __restrict__ wout,
    const float* __restrict__ bout,     // scalar
    float* __restrict__ adj_out,        // (B, 121)
    float* __restrict__ az_out,         // (B, 11)
    float* __restrict__ basem_out)      // (B, 512)
{
  extern __shared__ __align__(64) char smem[];
  _Float16* P     = (_Float16*)smem;            // 11 x 512 f16
  _Float16* H1T   = P + 11 * NF;                // 121 x 512 f16 (transposed H1)
  _Float16* panel = H1T + NP * NF;              // 16 x 512 f16 weight panel
  float*    adjl  = (float*)(panel + 16 * NF);  // 128 f32
  float*    azl   = adjl + NPPAD;               // 16 f32

  const int b    = blockIdx.x;
  const int tid  = threadIdx.x;
  const int lane = tid & 31;
  const int wave = tid >> 5;
  const int p0   = wave * 16;

  // Stage P (f16): rows 0..9 = nf, row 10 = current col.
  for (int idx = tid; idx < 11 * NF; idx += 256) {
    int n = idx >> 9, f = idx & (NF - 1);
    float v = (n < 10) ? nf[(size_t)b * NN * NF + n * NF + f]
                       : hcol[(size_t)b * NF + f];
    P[idx] = (_Float16)v;
  }
  __syncthreads();

  // E-tile B-fragments for all 16 k-steps (K = 512), resident in VGPRs.
  v16h bf[16];
  {
    const int col  = lane & 15;
    const int koff = (lane >> 4) * 16;
    const int p    = p0 + col;
    const bool valid = (p < NP);
    const int i = valid ? (p / NN) : 0;
    const int j = valid ? (p % NN) : 0;
#pragma unroll
    for (int k = 0; k < 16; ++k) {
      if (valid) {
        v16h pi = *(const v16h*)(P + i * NF + k * 32 + koff);
        v16h pj = *(const v16h*)(P + j * NF + k * 32 + koff);
        bf[k] = pi * pj;                 // packed f16 multiply
      } else {
        v16h z = {};
        bf[k] = z;
      }
    }
  }

  // ---- GEMM1: H1 = relu(W1 * E + b1) -> store transposed H1T[p][g] (f16).
  for (int gt = 0; gt < 32; ++gt) {
    stage_panel(panel, W1h + (size_t)gt * 16 * NF, tid);
    __syncthreads();
    v8f acc = {};
#pragma unroll
    for (int k = 0; k < 16; ++k)
      acc = wmma_f16(load_a_frag(panel + k * 32, NF, lane), bf[k], acc);
    const int col = lane & 15, kh = lane >> 4;
    const int p = p0 + col;
    if (p < NP) {
#pragma unroll
      for (int r = 0; r < 8; ++r) {
        int g = gt * 16 + r + kh * 8;
        float v = acc[r] + b1[g];
        v = v > 0.f ? v : 0.f;
        H1T[p * NF + g] = (_Float16)v;
      }
    }
    __syncthreads();    // panel consumed + H1T slice visible
  }

  // Reload bf[] with GEMM2 B-fragments (H1T columns; K dim = g).
  // Pad columns p>=121 read in-allocation garbage -> feeds only dropped outputs.
  {
    const int col  = lane & 15;
    const int koff = (lane >> 4) * 16;
    const int p    = p0 + col;
#pragma unroll
    for (int k = 0; k < 16; ++k)
      bf[k] = *(const v16h*)(H1T + p * NF + k * 32 + koff);
  }

  // ---- GEMM2 + w_out reduction: psum[p] = sum_g wout[g]*relu(W2.H1 + b2)[g][p]
  float psum = 0.f;
  for (int gt = 0; gt < 32; ++gt) {
    stage_panel(panel, W2h + (size_t)gt * 16 * NF, tid);
    __syncthreads();
    v8f acc = {};
#pragma unroll
    for (int k = 0; k < 16; ++k)
      acc = wmma_f16(load_a_frag(panel + k * 32, NF, lane), bf[k], acc);
    const int kh = lane >> 4;
#pragma unroll
    for (int r = 0; r < 8; ++r) {
      int g = gt * 16 + r + kh * 8;
      float v = acc[r] + b2[g];
      v = v > 0.f ? v : 0.f;
      psum += v * wout[g];
    }
    __syncthreads();
  }
  psum += __shfl_xor(psum, 16);                 // fold the two M-halves (same p)
  {
    const int p = p0 + (lane & 15);
    if (lane < 16) adjl[p] = psum;
  }
  __syncthreads();

  // adj output (b_out is a uniform shift -> irrelevant for the softmax).
  if (tid < NP) adj_out[(size_t)b * NP + tid] = adjl[tid] + bout[0];

  // softmax over row i = 10 (p = 110..120) -> a_z
  if (tid == 0) {
    float m = -1e30f;
    for (int j = 0; j < NN; ++j) m = fmaxf(m, adjl[110 + j]);
    float e[NN], s = 0.f;
    for (int j = 0; j < NN; ++j) { e[j] = __expf(adjl[110 + j] - m); s += e[j]; }
    float inv = 1.f / s;
    for (int j = 0; j < NN; ++j) {
      azl[j] = e[j] * inv;
      az_out[(size_t)b * NN + j] = azl[j];
    }
  }
  __syncthreads();

  // base_m[f] = sum_{j<10} a_z[j] * nf[b][j][f]  (fp32 from global for accuracy)
  for (int f = tid; f < NF; f += 256) {
    float s = 0.f;
#pragma unroll
    for (int j = 0; j < 10; ++j)
      s += azl[j] * nf[(size_t)b * NN * NF + j * NF + f];
    basem_out[(size_t)b * NF + f] = s;
  }
}

// ---------------------------------------------------------------------------
// GRU kernel (one e-step): m_z = base_m + a_z[10]*h_col;
// gi = m_z . W_ih^T, gh = h_col . W_hh^T (16b x 1536 x 512 WMMA GEMMs);
// o-tiles are partitioned across waves, so each weight row is read from L2
// exactly once per block. Gate math in fp32; h_col updated in place.
// ---------------------------------------------------------------------------
__global__ __launch_bounds__(256) void gru_kernel(
    const float* __restrict__ basem,   // (B, 512)
    const float* __restrict__ az,      // (B, 11)
    float* __restrict__ hcol,          // (B, 512) in/out
    const _Float16* __restrict__ Wih,  // (1536, 512) f16
    const _Float16* __restrict__ Whh)  // (1536, 512) f16
{
  extern __shared__ __align__(64) char smem[];
  _Float16* Mz    = (_Float16*)smem;             // 16 x 512 f16
  _Float16* Hc    = Mz + 16 * NF;                // 16 x 512 f16
  float*    sumrz = (float*)(Hc + 16 * NF);      // 16 x 1024 f32 (r,z pre-act)
  float*    gin   = sumrz + 16 * 1024;           // 16 x 512 f32 (i_n)
  float*    ghn   = gin + 16 * NF;               // 16 x 512 f32 (h_n)

  const int b0   = blockIdx.x * 16;
  const int tid  = threadIdx.x;
  const int lane = tid & 31;
  const int wave = tid >> 5;

  for (int idx = tid; idx < 16 * NF; idx += 256) {
    int bl = idx >> 9, f = idx & (NF - 1);
    int b = b0 + bl;
    float h = hcol[(size_t)b * NF + f];
    float m = basem[(size_t)b * NF + f] + az[(size_t)b * NN + 10] * h;
    Mz[bl * NF + f] = (_Float16)m;
    Hc[bl * NF + f] = (_Float16)h;
  }
  __syncthreads();

  for (int t = 0; t < 12; ++t) {
    const int ot = wave * 12 + t;       // 96 o-tiles cover 1536 outputs
    const int o0 = ot * 16;
    v8f ai = {}, ah = {};
#pragma unroll
    for (int k = 0; k < 16; ++k) {
      v16h am = load_a_frag(Mz + k * 32, NF, lane);
      v16h av = load_a_frag(Hc + k * 32, NF, lane);
      v16h bi = load_bt_frag(Wih + (size_t)o0 * NF + k * 32, NF, lane);
      v16h bh = load_bt_frag(Whh + (size_t)o0 * NF + k * 32, NF, lane);
      ai = wmma_f16(am, bi, ai);
      ah = wmma_f16(av, bh, ah);
    }
    const int col = lane & 15, kh = lane >> 4;
    const int o = o0 + col;
#pragma unroll
    for (int r = 0; r < 8; ++r) {
      int bl = r + kh * 8;
      if (o < 1024) {
        sumrz[bl * 1024 + o] = ai[r] + ah[r];      // r/z gates need gi+gh only
      } else {
        gin[bl * NF + (o - 1024)] = ai[r];
        ghn[bl * NF + (o - 1024)] = ah[r];
      }
    }
  }
  __syncthreads();

  for (int idx = tid; idx < 16 * NF; idx += 256) {
    int bl = idx >> 9, f = idx & (NF - 1);
    int b = b0 + bl;
    float hprev = hcol[(size_t)b * NF + f];
    float r = 1.f / (1.f + __expf(-sumrz[bl * 1024 + f]));
    float z = 1.f / (1.f + __expf(-sumrz[bl * 1024 + 512 + f]));
    float n = tanhf(gin[bl * NF + f] + r * ghn[bl * NF + f]);
    hcol[(size_t)b * NF + f] = (1.f - z) * n + z * hprev;
  }
}

// ---------------------------------------------------------------------------
// Host-side orchestration (all launches on `stream`, graph-capture safe).
// ---------------------------------------------------------------------------
extern "C" void kernel_launch(void* const* d_in, const int* in_sizes, int n_in,
                              void* d_out, int out_size, void* d_ws, size_t ws_size,
                              hipStream_t stream) {
  const float* nf   = (const float*)d_in[0];  // node_features (1024,11,512)
  const float* W1   = (const float*)d_in[1];  // (512,512)
  const float* b1   = (const float*)d_in[2];  // (512,)
  const float* W2   = (const float*)d_in[3];  // (512,512)
  const float* b2   = (const float*)d_in[4];  // (512,)
  const float* wout = (const float*)d_in[5];  // (512,)
  const float* bout = (const float*)d_in[6];  // scalar
  const float* Wih  = (const float*)d_in[7];  // (1536,512)
  const float* Whh  = (const float*)d_in[8];  // (1536,512)

  // Workspace carve-out (~8.4 MB total).
  char* ws = (char*)d_ws;
  size_t off = 0;
  auto carve = [&](size_t bytes) -> void* {
    void* p = ws + off;
    off = (off + bytes + 255) & ~(size_t)255;
    return p;
  };
  _Float16* W1h  = (_Float16*)carve((size_t)512 * 512 * 2);
  _Float16* W2h  = (_Float16*)carve((size_t)512 * 512 * 2);
  _Float16* Wihh = (_Float16*)carve((size_t)1536 * 512 * 2);
  _Float16* Whhh = (_Float16*)carve((size_t)1536 * 512 * 2);
  float* hcol    = (float*)carve((size_t)NB * NF * 4);
  float* basem   = (float*)carve((size_t)NB * NF * 4);
  float* az      = (float*)carve((size_t)NB * NN * 4);

  float* adj_out = (float*)d_out;                 // (1024, 121)
  float* col_out = adj_out + (size_t)NB * NP;     // (1024, 512)

  // Weight conversion fp32 -> f16, and h_col init from x0[:,:,zidx].
  cvt_f16_kernel<<<(512 * 512 + 255) / 256, 256, 0, stream>>>(W1, W1h, 512 * 512);
  cvt_f16_kernel<<<(512 * 512 + 255) / 256, 256, 0, stream>>>(W2, W2h, 512 * 512);
  cvt_f16_kernel<<<(1536 * 512 + 255) / 256, 256, 0, stream>>>(Wih, Wihh, 1536 * 512);
  cvt_f16_kernel<<<(1536 * 512 + 255) / 256, 256, 0, stream>>>(Whh, Whhh, 1536 * 512);
  init_col_kernel<<<(NB * NF + 255) / 256, 256, 0, stream>>>(nf, hcol, NB * NF);

  const size_t adj_lds = (size_t)11 * NF * 2      // P
                       + (size_t)NP * NF * 2      // H1T
                       + (size_t)16 * NF * 2      // weight panel
                       + (NPPAD + 16) * 4;        // adjl + azl  => 152,128 B
  const size_t gru_lds = (size_t)16 * NF * 2 * 2
                       + (size_t)16 * 1024 * 4
                       + (size_t)16 * NF * 4 * 2; // 160 KB

  for (int t = 0; t < 3; ++t) {                                  // M_STEP
    adj_kernel<<<NB, 256, adj_lds, stream>>>(nf, hcol, W1h, W2h, b1, b2,
                                             wout, bout, adj_out, az, basem);
    for (int s = 0; s < 3; ++s)                                  // E_STEP
      gru_kernel<<<NB / 16, 256, gru_lds, stream>>>(basem, az, hcol, Wihh, Whhh);
  }

  copy_col_kernel<<<(NB * NF + 255) / 256, 256, 0, stream>>>(hcol, col_out, NB * NF);
}
